// BWGNN_31602369364073
// MI455X (gfx1250) — compile-verified, matrix operand order checked by reading, not run on hardware
//
#include <hip/hip_runtime.h>

typedef __attribute__((ext_vector_type(16))) _Float16 v16h;
typedef __attribute__((ext_vector_type(8)))  float    v8f;

#define FIN 128
#define HID 64

// ---------------- utility kernels ----------------

__global__ void fill_zero_kernel(float* p, long n) {
  long i = (long)blockIdx.x * blockDim.x + threadIdx.x;
  if (i < n) p[i] = 0.0f;
}

__global__ void degree_kernel(const int* __restrict__ src, float* __restrict__ deg, int E) {
  int e = blockIdx.x * blockDim.x + threadIdx.x;
  if (e < E) atomicAdd(&deg[src[e]], 1.0f);
}

__global__ void dinv_kernel(const float* __restrict__ deg, float* __restrict__ dinv, int N) {
  int n = blockIdx.x * blockDim.x + threadIdx.x;
  if (n < N) {
    float d = deg[n];
    d = d < 1.0f ? 1.0f : d;
    dinv[n] = rsqrtf(d);
  }
}

// Pack W (128x64 f32 row-major) into the CDNA5 16-bit B-fragment layout (f16).
// Fragment (s,t): 32 lanes x 16 halfs, contiguous per lane.
// lane L: n = t*16 + (L&15); base = (L>>4)*8;
//   j<8  -> K = s*32 + base + j
//   j>=8 -> K = s*32 + 16 + base + (j-8)
__global__ void packW_kernel(const float* __restrict__ W, _Float16* __restrict__ packedW) {
  int tid = threadIdx.x;  // 256 threads, 1 block; 8192 elements total
  for (int it = 0; it < 32; ++it) {
    int idx  = tid * 32 + it;
    int j    = idx & 15;
    int lane = (idx >> 4) & 31;
    int t    = (idx >> 9) & 3;
    int s    = (idx >> 11) & 3;
    int n    = t * 16 + (lane & 15);
    int base = (lane >> 4) * 8;
    int k    = s * 32 + (j < 8 ? (base + j) : (16 + base + (j - 8)));
    packedW[idx] = (_Float16)W[k * HID + n];
  }
}

// ---------------- WMMA projection: h = leaky_relu(feature @ W + b) ----------------
// One wave (32 threads) per 16-row tile; computes full 16x64 output, K=128 in 4 steps.
// Writes into d_out column block 0 (row stride 256).
__global__ void __launch_bounds__(32)
gemm_kernel(const float* __restrict__ feat, const _Float16* __restrict__ packedW,
            const float* __restrict__ bias, float* __restrict__ out) {
  int tile = blockIdx.x;
  int lane = threadIdx.x & 31;
  int m0   = tile * 16;
  int m    = m0 + (lane & 15);
  int halfsel = lane >> 4;

  v8f acc0 = {}, acc1 = {}, acc2 = {}, acc3 = {};
  const float* arow = feat + (long)m * FIN;
  const v16h*  bw   = (const v16h*)packedW;

#pragma unroll
  for (int s = 0; s < 4; ++s) {
    const float4* p0 = (const float4*)(arow + s * 32 + halfsel * 8);
    const float4* p1 = (const float4*)(arow + s * 32 + 16 + halfsel * 8);
    float4 f0 = p0[0], f1 = p0[1], f2 = p1[0], f3 = p1[1];
    v16h a;
    a[0]  = (_Float16)f0.x; a[1]  = (_Float16)f0.y; a[2]  = (_Float16)f0.z; a[3]  = (_Float16)f0.w;
    a[4]  = (_Float16)f1.x; a[5]  = (_Float16)f1.y; a[6]  = (_Float16)f1.z; a[7]  = (_Float16)f1.w;
    a[8]  = (_Float16)f2.x; a[9]  = (_Float16)f2.y; a[10] = (_Float16)f2.z; a[11] = (_Float16)f2.w;
    a[12] = (_Float16)f3.x; a[13] = (_Float16)f3.y; a[14] = (_Float16)f3.z; a[15] = (_Float16)f3.w;

    v16h b0 = bw[(s * 4 + 0) * 32 + lane];
    v16h b1 = bw[(s * 4 + 1) * 32 + lane];
    v16h b2 = bw[(s * 4 + 2) * 32 + lane];
    v16h b3 = bw[(s * 4 + 3) * 32 + lane];
    acc0 = __builtin_amdgcn_wmma_f32_16x16x32_f16(false, a, false, b0, (short)0, acc0, false, false);
    acc1 = __builtin_amdgcn_wmma_f32_16x16x32_f16(false, a, false, b1, (short)0, acc1, false, false);
    acc2 = __builtin_amdgcn_wmma_f32_16x16x32_f16(false, a, false, b2, (short)0, acc2, false, false);
    acc3 = __builtin_amdgcn_wmma_f32_16x16x32_f16(false, a, false, b3, (short)0, acc3, false, false);
  }

  // C/D layout: vgpr r, lanes 0-15 -> (M=r, N=lane); lanes 16-31 -> (M=8+r, N=lane-16)
  int col0 = lane & 15;
  int rowb = m0 + 8 * halfsel;
#define STORE_TILE(ACC, T)                                         \
  {                                                                \
    int col = (T) * 16 + col0;                                     \
    float bb = bias[col];                                          \
    _Pragma("unroll") for (int r = 0; r < 8; ++r) {                \
      float v = (ACC)[r] + bb;                                     \
      v = v > 0.0f ? v : 0.01f * v;                                \
      out[(long)(rowb + r) * 256 + col] = v;                       \
    }                                                              \
  }
  STORE_TILE(acc0, 0)
  STORE_TILE(acc1, 1)
  STORE_TILE(acc2, 2)
  STORE_TILE(acc3, 3)
#undef STORE_TILE
}

// ---------------- Laplacian message passing ----------------
// 16 threads per edge, float4 each: agg[dst] += P[src] * dinv[src]
__global__ void scatter_kernel(const int* __restrict__ src, const int* __restrict__ dst,
                               const float* __restrict__ dinv, const float* __restrict__ pin,
                               float* __restrict__ agg, int E) {
  long gid = (long)blockIdx.x * blockDim.x + threadIdx.x;
  int e   = (int)(gid >> 4);
  int sub = (int)(gid & 15);
  if (e >= E) return;
  int s = src[e];
  int d = dst[e];
  float ds = dinv[s];
  float4 v = ((const float4*)(pin + (long)s * 256))[sub];
  float* ap = agg + (long)d * 64 + sub * 4;
  atomicAdd(ap + 0, v.x * ds);
  atomicAdd(ap + 1, v.y * ds);
  atomicAdd(ap + 2, v.z * ds);
  atomicAdd(ap + 3, v.w * ds);
}

// P_new = P_old - dinv * agg   (P blocks live in d_out, row stride 256)
__global__ void apply_kernel(const float* __restrict__ dinv, const float* __restrict__ agg,
                             const float* __restrict__ pin, float* __restrict__ pout, long NH) {
  long i = (long)blockIdx.x * blockDim.x + threadIdx.x;
  if (i >= NH) return;
  int n = (int)(i >> 6);
  int f = (int)(i & 63);
  pout[(long)n * 256 + f] = pin[(long)n * 256 + f] - dinv[n] * agg[i];
}

// In-place 4x4 mix: out_i = sum_k Theta[i][k] * P_k  (Theta from calculate_theta2(4))
__global__ void mix_kernel(float* __restrict__ out, long NH) {
  long i = (long)blockIdx.x * blockDim.x + threadIdx.x;
  if (i >= NH) return;
  int n = (int)(i >> 6);
  int f = (int)(i & 63);
  float* row = out + (long)n * 256 + f;
  float p0 = row[0], p1 = row[64], p2 = row[128], p3 = row[192];
  row[0]   = 2.5f * p0 - 5.0f * p1 + 3.75f * p2 - 1.25f * p3;
  row[64]  =             5.0f * p1 - 7.5f  * p2 + 3.75f * p3;
  row[128] =                         3.75f * p2 - 3.75f * p3;
  row[192] =                                      1.25f * p3;
}

// ---------------- launch ----------------

extern "C" void kernel_launch(void* const* d_in, const int* in_sizes, int n_in,
                              void* d_out, int out_size, void* d_ws, size_t ws_size,
                              hipStream_t stream) {
  const float* feature = (const float*)d_in[0];
  const int*   ei      = (const int*)d_in[1];
  const float* W       = (const float*)d_in[2];
  const float* b       = (const float*)d_in[3];
  float*       out     = (float*)d_out;

  int N = in_sizes[0] / FIN;
  int E = in_sizes[1] / 2;
  const int* src = ei;
  const int* dst = ei + E;

  // ws layout: deg[N] | dinv[N] | agg[N*64] | packedW (8192 f16)
  float* ws   = (float*)d_ws;
  float* deg  = ws;
  float* dinv = ws + N;
  float* agg  = ws + 2L * N;
  _Float16* packedW = (_Float16*)(ws + 2L * N + (long)N * 64);

  long NH = (long)N * 64;

  fill_zero_kernel<<<(N + 255) / 256, 256, 0, stream>>>(deg, N);
  degree_kernel<<<(E + 255) / 256, 256, 0, stream>>>(src, deg, E);
  dinv_kernel<<<(N + 255) / 256, 256, 0, stream>>>(deg, dinv, N);
  packW_kernel<<<1, 256, 0, stream>>>(W, packedW);

  int ntiles = N / 16;  // N = 100000 -> 6250 tiles
  gemm_kernel<<<ntiles, 32, 0, stream>>>(feature, packedW, b, out);

  unsigned egrid = (unsigned)(((long)E * 16 + 255) / 256);
  unsigned ngrid = (unsigned)((NH + 255) / 256);
  for (int k = 1; k <= 3; ++k) {
    fill_zero_kernel<<<ngrid, 256, 0, stream>>>(agg, NH);
    scatter_kernel<<<egrid, 256, 0, stream>>>(src, dst, dinv, out + (k - 1) * 64, agg, E);
    apply_kernel<<<ngrid, 256, 0, stream>>>(dinv, agg, out + (k - 1) * 64, out + k * 64, NH);
  }
  mix_kernel<<<ngrid, 256, 0, stream>>>(out, NH);
}
